// CausalMultiHeadSelfAttention_73083163509573
// MI455X (gfx1250) — matrix operand order repaired and checked by reading, not compiled, not run on hardware
//
#include <hip/hip_runtime.h>

// ---------------------------------------------------------------------------
// Causal MHA for MI455X (gfx1250): bf16 WMMA (v_wmma_f32_16x16x32_bf16)
// pipeline: cvt -> QKV GEMM (async-LDS staged B) -> RoPE -> flash-attn ->
//           out GEMM (async-LDS staged B)
// ---------------------------------------------------------------------------

typedef __attribute__((ext_vector_type(16))) __bf16 v16bf;
typedef __attribute__((ext_vector_type(8)))  float  v8f;

#define D_MODEL 1024
#define N_HEADS 16
#define D_HEAD  64
#define SEQ     2048
#define BATCH   4
#define NTOK    (BATCH * SEQ)        // 8192 rows
#define NBH     (BATCH * N_HEADS)    // 64 (batch*head)

__device__ __forceinline__ v8f wmma_bf16(v16bf a, v16bf b, v8f c) {
  // 8 args: (neg_a, A, neg_b, B, c_mod, C, reuse_a, reuse_b)
  return __builtin_amdgcn_wmma_f32_16x16x32_bf16(false, a, false, b, (short)0, c,
                                                 false, false);
}

// A-fragment (16x32 bf16, M x K). Lane l holds row M = l&15.
// lanes 0-15:  K = kb+0..15
// lanes 16-31: K = kb+16..31
__device__ __forceinline__ v16bf load_a_frag(const __bf16* __restrict__ rowp,
                                             int kb, int sel) {
  const __bf16* p0 = rowp + kb + sel * 16;
  const __bf16* p1 = rowp + kb + sel * 16 + 8;
  v16bf a;
#pragma unroll
  for (int i = 0; i < 8; ++i) {
    a[i]     = p0[i];
    a[8 + i] = p1[i];
  }
  return a;
}

// row-wise reductions across the 16-lane half-groups of a wave32
__device__ __forceinline__ float rowmax16(float v) {
#pragma unroll
  for (int off = 8; off >= 1; off >>= 1) v = fmaxf(v, __shfl_xor(v, off, 32));
  return v;
}
__device__ __forceinline__ float rowsum16(float v) {
#pragma unroll
  for (int off = 8; off >= 1; off >>= 1) v += __shfl_xor(v, off, 32);
  return v;
}

// ---- CDNA5 async global->LDS copy (ASYNCcnt-tracked) ----------------------
// dst operand = VGPR with LDS byte address (low 32 bits of generic pointer),
// src = 64-bit global address.
__device__ __forceinline__ void async_g2l_b128(void* lds_ptr, const void* gptr) {
  unsigned int l = (unsigned int)(unsigned long long)(uintptr_t)lds_ptr;
  unsigned long long g = (unsigned long long)(uintptr_t)gptr;
  asm volatile("global_load_async_to_lds_b128 %0, %1, off"
               :: "v"(l), "v"(g) : "memory");
}
__device__ __forceinline__ void wait_async_le1() {
  asm volatile("s_wait_asynccnt 0x1" ::: "memory");
}
__device__ __forceinline__ void wait_async_0() {
  asm volatile("s_wait_asynccnt 0x0" ::: "memory");
}

// Stage one 64-row x 32-col bf16 B panel (4 KB) block-cooperatively:
// 256 threads x 16 bytes, one async instruction per thread.
__device__ __forceinline__ void stage_panelB(const __bf16* __restrict__ W,
                                             int n0, int kb,
                                             __bf16 (*dst)[32], int tid) {
  int r = tid >> 2;        // 0..63  (row within panel)
  int c = tid & 3;         // 0..3   (16-byte chunk = 8 bf16)
  const __bf16* g = W + (size_t)(n0 + r) * D_MODEL + kb + c * 8;
  async_g2l_b128(&dst[r][c * 8], g);
}

// ---------------------------------------------------------------------------
// fp32 -> bf16 one-time conversion
// ---------------------------------------------------------------------------
__global__ __launch_bounds__(256) void cvt_f32_bf16_kernel(
    const float* __restrict__ in, __bf16* __restrict__ out, int n) {
  int i = blockIdx.x * 256 + threadIdx.x;
  if (i < n) out[i] = (__bf16)in[i];
}

// ---------------------------------------------------------------------------
// QKV projection: [8192,1024]bf16 x [3072,1024]^T bf16 -> scatter to
// Q[b,h,s,64], K[b,h,s,64], Vt[b,h,64,s].
// grid (64, 48), block 256 (8 waves). Wave = 16 rows x 64 cols.
// B panel double-buffered in LDS via async copies.
// ---------------------------------------------------------------------------
__global__ __launch_bounds__(256) void qkv_gemm_kernel(
    const __bf16* __restrict__ X, const __bf16* __restrict__ W,
    __bf16* __restrict__ Q, __bf16* __restrict__ K, __bf16* __restrict__ Vt) {
  __shared__ __align__(64) __bf16 panelB[2][64][32];  // 2 x 4 KB

  const int tid  = threadIdx.x;
  const int lane = tid & 31;
  const int wave = tid >> 5;
  const int lrow = lane & 15;
  const int sel  = lane >> 4;
  const int m0 = (blockIdx.x * 8 + wave) * 16;
  const int n0 = blockIdx.y * 64;

  const __bf16* arow = X + (size_t)(m0 + lrow) * D_MODEL;
  v8f acc[4] = {};

  stage_panelB(W, n0, 0, panelB[0], tid);

  for (int kb = 0; kb < D_MODEL; kb += 32) {
    const int cur = (kb >> 5) & 1;
    const bool more = (kb + 32) < D_MODEL;
    if (more) stage_panelB(W, n0, kb + 32, panelB[cur ^ 1], tid);
    if (more) wait_async_le1(); else wait_async_0();
    __syncthreads();  // panel[cur] visible from all threads

    v16bf a = load_a_frag(arow, kb, sel);
#pragma unroll
    for (int t = 0; t < 4; ++t) {
      // B-frag (32x16): lane covers column n0+t*16+lrow, K run of 16 at sel*16
      v16bf bfr = *(const v16bf*)&panelB[cur][t * 16 + lrow][sel * 16];
      acc[t] = wmma_bf16(a, bfr, acc[t]);
    }
    __syncthreads();  // all reads of panel[cur] done before it is refilled
  }

#pragma unroll
  for (int t = 0; t < 4; ++t) {
#pragma unroll
    for (int r = 0; r < 8; ++r) {
      // C layout: VGPR r -> M = r (lanes 0-15), M = r+8 (lanes 16-31); N = lane%16
      int gr   = m0 + r + 8 * sel;
      int b    = gr >> 11;
      int spos = gr & (SEQ - 1);
      int e    = n0 + t * 16 + lrow;
      __bf16 bv = (__bf16)acc[t][r];
      int seg = e >> 10;  // 0:q 1:k 2:v
      int ehd = e & 1023;
      int h = ehd >> 6, d = ehd & 63;
      size_t bh = (size_t)b * N_HEADS + h;
      if (seg == 0)      Q [(bh * SEQ + spos) * D_HEAD + d] = bv;
      else if (seg == 1) K [(bh * SEQ + spos) * D_HEAD + d] = bv;
      else               Vt[(bh * D_HEAD + d) * SEQ + spos] = bv;
    }
  }
}

// ---------------------------------------------------------------------------
// RoPE in place on Q and K; folds softmax scale (1/8) into Q.
// ---------------------------------------------------------------------------
__global__ __launch_bounds__(256) void rope_kernel(__bf16* __restrict__ Q,
                                                   __bf16* __restrict__ K) {
  size_t idx = (size_t)blockIdx.x * 256 + threadIdx.x;  // < 8388608
  int i = (int)(idx & 31);  // rotation pair index (half = 32)
  size_t rest = idx >> 5;
  int spos = (int)(rest & (SEQ - 1));
  rest >>= 11;
  int h = (int)(rest & (N_HEADS - 1));
  rest >>= 4;
  int b  = (int)(rest & (BATCH - 1));
  int qk = (int)(rest >> 2);

  __bf16* base = qk ? K : Q;
  size_t off = ((((size_t)b * N_HEADS + h) * SEQ) + spos) * D_HEAD + 2 * i;
  float x1 = (float)base[off];
  float x2 = (float)base[off + 1];
  float freq = __expf(-(float)(2 * i) * (1.0f / 64.0f) * 9.210340371976184f);
  float ang = (float)spos * freq;
  float s, c;
  __sincosf(ang, &s, &c);
  float o1 = x1 * c - x2 * s;
  float o2 = x1 * s + x2 * c;
  if (qk == 0) { o1 *= 0.125f; o2 *= 0.125f; }  // 1/sqrt(D_HEAD)
  base[off]     = (__bf16)o1;
  base[off + 1] = (__bf16)o2;
}

// ---------------------------------------------------------------------------
// Flash attention (causal, online softmax). grid (16, 64), block 256 (8 waves).
// wave = 16 query rows; 32-key chunks; S = 4 WMMA, P*V = 4 WMMA per chunk.
// ---------------------------------------------------------------------------
__global__ __launch_bounds__(256) void attn_kernel(
    const __bf16* __restrict__ Q, const __bf16* __restrict__ K,
    const __bf16* __restrict__ Vt, __bf16* __restrict__ AO) {
  __shared__ __bf16 ldsP[8][16][32];  // per-wave 16x32 bf16 P staging

  const int lane = threadIdx.x & 31;
  const int wave = threadIdx.x >> 5;
  const int lrow = lane & 15;
  const int sel  = lane >> 4;
  const int bh = blockIdx.y;
  const int b  = bh >> 4;
  const int h  = bh & 15;
  const int qbase = blockIdx.x * 128 + wave * 16;

  const __bf16* Qh  = Q  + (size_t)bh * SEQ * D_HEAD;
  const __bf16* Kh  = K  + (size_t)bh * SEQ * D_HEAD;
  const __bf16* Vth = Vt + (size_t)bh * D_HEAD * SEQ;

  v16bf qa[2];
  {
    const __bf16* qrow = Qh + (size_t)(qbase + lrow) * D_HEAD;
#pragma unroll
    for (int kk = 0; kk < 2; ++kk) qa[kk] = load_a_frag(qrow, kk * 32, sel);
  }

  float m[8], l[8];
  v8f o[4] = {};
#pragma unroll
  for (int r = 0; r < 8; ++r) { m[r] = -1e30f; l[r] = 0.0f; }

  for (int kb = 0; kb <= qbase + 15; kb += 32) {
    // ---- S = Q * K^T for 32 keys (two 16-col C-frags) ----
    v8f sc[2];
#pragma unroll
    for (int t = 0; t < 2; ++t) {
      v8f a = {};
#pragma unroll
      for (int kk = 0; kk < 2; ++kk) {
        const __bf16* kp =
            Kh + (size_t)(kb + t * 16 + lrow) * D_HEAD + kk * 32 + sel * 16;
        v16bf bfr = *(const v16bf*)kp;
        a = wmma_bf16(qa[kk], bfr, a);
      }
      sc[t] = a;
    }

    // ---- causal mask + online softmax ----
    float corr[8];
#pragma unroll
    for (int r = 0; r < 8; ++r) {
      int row  = qbase + r + 8 * sel;
      int key0 = kb + lrow;
      int key1 = kb + 16 + lrow;
      float v0 = (key0 <= row) ? sc[0][r] : -1e30f;
      float v1 = (key1 <= row) ? sc[1][r] : -1e30f;
      float mx = rowmax16(fmaxf(v0, v1));
      float mn = fmaxf(m[r], mx);
      float p0 = __expf(v0 - mn);
      float p1 = __expf(v1 - mn);
      float sum = rowsum16(p0 + p1);
      float c = __expf(m[r] - mn);
      l[r] = l[r] * c + sum;
      m[r] = mn;
      corr[r] = c;
      int prow = r + 8 * sel;
      ldsP[wave][prow][lrow]      = (__bf16)p0;
      ldsP[wave][prow][16 + lrow] = (__bf16)p1;
    }

#pragma unroll
    for (int t = 0; t < 4; ++t) {
#pragma unroll
      for (int r = 0; r < 8; ++r) o[t][r] = o[t][r] * corr[r];
    }

    // ---- rebuild P as an A-frag from LDS (same-wave DS ops are in-order) ----
    v16bf pa;
#pragma unroll
    for (int i = 0; i < 8; ++i) {
      pa[i]     = ldsP[wave][lrow][sel * 16 + i];
      pa[8 + i] = ldsP[wave][lrow][sel * 16 + 8 + i];
    }

    // ---- O += P * V ----
#pragma unroll
    for (int t = 0; t < 4; ++t) {
      const __bf16* vp = Vth + (size_t)(t * 16 + lrow) * SEQ + kb + sel * 16;
      v16bf bfr = *(const v16bf*)vp;
      o[t] = wmma_bf16(pa, bfr, o[t]);
    }
  }

#pragma unroll
  for (int t = 0; t < 4; ++t) {
#pragma unroll
    for (int r = 0; r < 8; ++r) {
      int row = qbase + r + 8 * sel;
      float val = o[t][r] / l[r];
      size_t off =
          ((size_t)b * SEQ + row) * D_MODEL + h * D_HEAD + t * 16 + lrow;
      AO[off] = (__bf16)val;
    }
  }
}

// ---------------------------------------------------------------------------
// Output projection: attn_out[8192,1024]bf16 x w_out[1024,1024]^T bf16 -> f32
// grid (64, 16), block 256.  Same async-LDS B staging as QKV GEMM.
// ---------------------------------------------------------------------------
__global__ __launch_bounds__(256) void out_gemm_kernel(
    const __bf16* __restrict__ A, const __bf16* __restrict__ W,
    float* __restrict__ out) {
  __shared__ __align__(64) __bf16 panelB[2][64][32];

  const int tid  = threadIdx.x;
  const int lane = tid & 31;
  const int wave = tid >> 5;
  const int lrow = lane & 15;
  const int sel  = lane >> 4;
  const int m0 = (blockIdx.x * 8 + wave) * 16;
  const int n0 = blockIdx.y * 64;

  const __bf16* arow = A + (size_t)(m0 + lrow) * D_MODEL;
  v8f acc[4] = {};

  stage_panelB(W, n0, 0, panelB[0], tid);

  for (int kb = 0; kb < D_MODEL; kb += 32) {
    const int cur = (kb >> 5) & 1;
    const bool more = (kb + 32) < D_MODEL;
    if (more) stage_panelB(W, n0, kb + 32, panelB[cur ^ 1], tid);
    if (more) wait_async_le1(); else wait_async_0();
    __syncthreads();

    v16bf a = load_a_frag(arow, kb, sel);
#pragma unroll
    for (int t = 0; t < 4; ++t) {
      v16bf bfr = *(const v16bf*)&panelB[cur][t * 16 + lrow][sel * 16];
      acc[t] = wmma_bf16(a, bfr, acc[t]);
    }
    __syncthreads();
  }

#pragma unroll
  for (int t = 0; t < 4; ++t) {
#pragma unroll
    for (int r = 0; r < 8; ++r) {
      int gr = m0 + r + 8 * sel;
      int e  = n0 + t * 16 + lrow;
      out[(size_t)gr * D_MODEL + e] = acc[t][r];
    }
  }
}

// ---------------------------------------------------------------------------
// launch
// ---------------------------------------------------------------------------
extern "C" void kernel_launch(void* const* d_in, const int* in_sizes, int n_in,
                              void* d_out, int out_size, void* d_ws,
                              size_t ws_size, hipStream_t stream) {
  const float* x     = (const float*)d_in[0];  // [4,2048,1024]
  const float* w_qkv = (const float*)d_in[1];  // [3072,1024]
  const float* w_out = (const float*)d_in[2];  // [1024,1024]
  float* out = (float*)d_out;                  // [4,2048,1024]

  char* ws = (char*)d_ws;
  const size_t MB = 1024 * 1024;
  __bf16* Xbf   = (__bf16*)(ws + 0 * MB);   // 16 MB
  __bf16* Wqkvb = (__bf16*)(ws + 16 * MB);  //  6 MB
  __bf16* Woutb = (__bf16*)(ws + 22 * MB);  //  2 MB
  __bf16* Qb    = (__bf16*)(ws + 24 * MB);  // 16 MB  [b,h,s,64]
  __bf16* Kb    = (__bf16*)(ws + 40 * MB);  // 16 MB  [b,h,s,64]
  __bf16* Vtb   = (__bf16*)(ws + 56 * MB);  // 16 MB  [b,h,64,s]
  __bf16* AOb   = (__bf16*)(ws + 72 * MB);  // 16 MB  [b,s,1024]

  const int nX  = NTOK * D_MODEL;          // 8388608
  const int nWq = 3 * D_MODEL * D_MODEL;   // 3145728
  const int nWo = D_MODEL * D_MODEL;       // 1048576

  cvt_f32_bf16_kernel<<<nX / 256, 256, 0, stream>>>(x, Xbf, nX);
  cvt_f32_bf16_kernel<<<nWq / 256, 256, 0, stream>>>(w_qkv, Wqkvb, nWq);
  cvt_f32_bf16_kernel<<<nWo / 256, 256, 0, stream>>>(w_out, Woutb, nWo);

  qkv_gemm_kernel<<<dim3(64, 48), 256, 0, stream>>>(Xbf, Wqkvb, Qb, Kb, Vtb);

  rope_kernel<<<8388608 / 256, 256, 0, stream>>>(Qb, Kb);

  attn_kernel<<<dim3(SEQ / 128, NBH), 256, 0, stream>>>(Qb, Kb, Vtb, AOb);

  out_gemm_kernel<<<dim3(64, 16), 256, 0, stream>>>(AOb, Woutb, out);
}
